// DCNBlock_35699768165204
// MI455X (gfx1250) — compile-verified, matrix operand order checked by reading, not run on hardware
//
#include <hip/hip_runtime.h>

// ---------------- types ----------------
typedef __attribute__((ext_vector_type(8)))  float  v8f;
typedef __attribute__((ext_vector_type(16))) __bf16 v16bf;
typedef __attribute__((ext_vector_type(4)))  int    v4i;

union V16 {
    v16bf v;
    unsigned short u[16];
};

#define Hc   128
#define Wc   128
#define HWc  (128 * 128)
#define Cc   64
#define CHWc (Cc * HWc)
#define Kc   9
#define KTOT 576          // C*K reduction length
#define TSTEPS 18         // 576 / 32

#define BMAIN_ELEMS (TSTEPS * 4 * 32 * 16)   // 36864 bf16 per hi/lo
#define BOFF_ELEMS  (TSTEPS * 2 * 32 * 16)   // 18432 bf16 per hi/lo
#define BMAIN_BYTES (BMAIN_ELEMS * 2)        // 73728
#define BOFF_BYTES  (BOFF_ELEMS * 2)         // 36864

// ------------- async global->LDS copy (CDNA5), with safe fallback ---------
#if defined(__has_builtin)
#if __has_builtin(__builtin_amdgcn_global_load_async_to_lds_b128) && \
    __has_builtin(__builtin_amdgcn_s_wait_asynccnt)
#define USE_ASYNC_LDS 1
#endif
#endif
#ifndef USE_ASYNC_LDS
#define USE_ASYNC_LDS 0
#endif

#define AS1P(p) ((__attribute__((address_space(1))) v4i*)(unsigned long long)(p))
#define AS3P(p) ((__attribute__((address_space(3))) v4i*)(unsigned int)(unsigned long long)(p))

static __device__ __forceinline__ void lds_fill_b128(char* dst, const char* src,
                                                     int bytes, int tid, int nthr) {
    int n16 = bytes >> 4;
    for (int i = tid; i < n16; i += nthr) {
#if USE_ASYNC_LDS
        __builtin_amdgcn_global_load_async_to_lds_b128(AS1P(src + (i << 4)),
                                                       AS3P(dst + (i << 4)), 0, 0);
#else
        ((uint4*)dst)[i] = ((const uint4*)src)[i];
#endif
    }
}

static __device__ __forceinline__ void lds_fill_wait() {
#if USE_ASYNC_LDS
    __builtin_amdgcn_s_wait_asynccnt(0);
#endif
    __syncthreads();
}

// ---------------- bf16 split helpers ----------------
static __device__ __forceinline__ unsigned short f2bf_rne(float f) {
    unsigned u = __float_as_uint(f);
    unsigned r = u + 0x7FFFu + ((u >> 16) & 1u);
    return (unsigned short)(r >> 16);
}
static __device__ __forceinline__ float bf2f(unsigned short h) {
    return __uint_as_float(((unsigned)h) << 16);
}

// ---------------------------------------------------------------------------
// Pack main conv weights w[o=64][c=64][3][3] into WMMA-B register layout,
// split into bf16 hi/lo.  Layout: [t=18][nt=4][lane=32][j=16] with
// reduction index r = t*32 + K', K' per 16-bit B 32x16 lane layout,
// r = k*64 + c, column N = nt*16 + lane%16.
// ---------------------------------------------------------------------------
__global__ void pack_w_main(const float* __restrict__ w,
                            unsigned short* __restrict__ bh,
                            unsigned short* __restrict__ bl) {
    int idx = blockIdx.x * blockDim.x + threadIdx.x;
    if (idx >= BMAIN_ELEMS) return;
    int j    = idx & 15;
    int lane = (idx >> 4) & 31;
    int nt   = (idx >> 9) & 3;
    int t    = idx >> 11;
    int N    = nt * 16 + (lane & 15);
    int base = (lane >> 4) * 8;
    int Kp   = base + (j < 8 ? j : j + 8);
    int r    = t * 32 + Kp;
    int k    = r >> 6;
    int c    = r & 63;
    float val = w[N * KTOT + c * Kc + k];
    unsigned short hi = f2bf_rne(val);
    bh[idx] = hi;
    bl[idx] = f2bf_rne(val - bf2f(hi));
}

// Offset conv weights w_off[18][64][3][3], columns padded 18 -> 32.
__global__ void pack_w_off(const float* __restrict__ w,
                           unsigned short* __restrict__ bh,
                           unsigned short* __restrict__ bl) {
    int idx = blockIdx.x * blockDim.x + threadIdx.x;
    if (idx >= BOFF_ELEMS) return;
    int j    = idx & 15;
    int lane = (idx >> 4) & 31;
    int nt   = (idx >> 9) & 1;
    int t    = idx >> 10;
    int N    = nt * 16 + (lane & 15);
    int base = (lane >> 4) * 8;
    int Kp   = base + (j < 8 ? j : j + 8);
    int r    = t * 32 + Kp;
    int k    = r >> 6;
    int c    = r & 63;
    float val = (N < 18) ? w[N * KTOT + c * Kc + k] : 0.0f;
    unsigned short hi = f2bf_rne(val);
    bh[idx] = hi;
    bl[idx] = f2bf_rne(val - bf2f(hi));
}

// Fold BN into scale/bias.
__global__ void prep_bn(const float* __restrict__ g, const float* __restrict__ be,
                        const float* __restrict__ m, const float* __restrict__ v,
                        float* __restrict__ sc, float* __restrict__ bi) {
    int o = threadIdx.x;
    if (o < Cc) {
        float inv = g[o] * rsqrtf(v[o] + 1e-5f);
        sc[o] = inv;
        bi[o] = be[o] - m[o] * inv;
    }
}

// ---------------------------------------------------------------------------
// Offset conv: 3x3, 64 -> 18 channels, zero padding.  One wave per
// 16-pixel row tile; im2col built in registers (unconditional clamped
// loads * 0/1 mask -> no EXEC divergence); B staged in LDS; bf16x3 WMMA.
// off layout: [b][18][H][W]
// ---------------------------------------------------------------------------
__global__ void __launch_bounds__(128)
offconv_kernel(const float* __restrict__ x,
               const unsigned short* __restrict__ bh,
               const unsigned short* __restrict__ bl,
               const float* __restrict__ boff,
               float* __restrict__ off) {
    extern __shared__ char smem[];
    char* lds_bh = smem;
    char* lds_bl = smem + BOFF_BYTES;

    lds_fill_b128(lds_bh, (const char*)bh, BOFF_BYTES, threadIdx.x, 128);
    lds_fill_b128(lds_bl, (const char*)bl, BOFF_BYTES, threadIdx.x, 128);
    lds_fill_wait();

    int lane  = threadIdx.x & 31;
    int tile  = blockIdx.x * 4 + (threadIdx.x >> 5);
    int b     = tile >> 10;           // 128 rows * 8 tiles = 1024 tiles/image
    int rem   = tile & 1023;
    int h     = rem >> 3;
    int w0    = (rem & 7) << 4;
    int p     = lane & 15;            // A-row (pixel) this lane feeds
    int kbase = (lane >> 4) * 8;      // lane-half K' offset

    const float* xb = x + (size_t)b * CHWc;
    v8f acc0 = {};
    v8f acc1 = {};

    for (int t = 0; t < TSTEPS; ++t) {
        int k  = t >> 1;
        int yy = h + (k / 3) - 1;
        int xx = w0 + p + (k % 3) - 1;
        float mask = ((yy >= 0) & (yy < Hc) & (xx >= 0) & (xx < Wc)) ? 1.0f : 0.0f;
        int yyc = min(max(yy, 0), Hc - 1);
        int xxc = min(max(xx, 0), Wc - 1);
        const float* xp = xb + yyc * Wc + xxc;
        int cbase = (t & 1) * 32;

        V16 ah, al;
#pragma unroll
        for (int j = 0; j < 16; ++j) {
            int Kp = kbase + (j < 8 ? j : j + 8);
            int c  = cbase + Kp;
            float v = xp[c * HWc] * mask;
            unsigned short hbit = f2bf_rne(v);
            ah.u[j] = hbit;
            al.u[j] = f2bf_rne(v - bf2f(hbit));
        }
#pragma unroll
        for (int nt = 0; nt < 2; ++nt) {
            const v16bf* bph = (const v16bf*)(lds_bh + (((((t * 2 + nt) * 32) + lane) << 4) << 1));
            const v16bf* bpl = (const v16bf*)(lds_bl + (((((t * 2 + nt) * 32) + lane) << 4) << 1));
            v16bf bhv = *bph;
            v16bf blv = *bpl;
            v8f& acc = nt ? acc1 : acc0;
            acc = __builtin_amdgcn_wmma_f32_16x16x32_bf16(false, ah.v, false, bhv, (short)0, acc, false, false);
            acc = __builtin_amdgcn_wmma_f32_16x16x32_bf16(false, ah.v, false, blv, (short)0, acc, false, false);
            acc = __builtin_amdgcn_wmma_f32_16x16x32_bf16(false, al.v, false, bhv, (short)0, acc, false, false);
        }
    }

    int n  = lane & 15;
    int mh = (lane >> 4) * 8;
#pragma unroll
    for (int nt = 0; nt < 2; ++nt) {
        int ch = nt * 16 + n;
        if (ch < 18) {
            float bias = boff[ch];
            v8f acc = nt ? acc1 : acc0;
            float* op = off + ((size_t)(b * 18 + ch)) * HWc + h * Wc + w0;
#pragma unroll
            for (int r = 0; r < 8; ++r) op[mh + r] = acc[r] + bias;
        }
    }
}

// ---------------------------------------------------------------------------
// Fused deformable sample + GEMM + BN + ReLU.
// One wave per 16-pixel tile; D = 16 pixels x 64 out channels.
// Bilinear geometry computed once per tap, shared across 64 channels.
// B (hi+lo) staged in LDS per workgroup.
// ---------------------------------------------------------------------------
__global__ void __launch_bounds__(128)
dcn_kernel(const float* __restrict__ x,
           const float* __restrict__ off,
           const unsigned short* __restrict__ bh,
           const unsigned short* __restrict__ bl,
           const float* __restrict__ sc,
           const float* __restrict__ bi,
           float* __restrict__ out) {
    extern __shared__ char smem[];
    char* lds_bh = smem;
    char* lds_bl = smem + BMAIN_BYTES;

    lds_fill_b128(lds_bh, (const char*)bh, BMAIN_BYTES, threadIdx.x, 128);
    lds_fill_b128(lds_bl, (const char*)bl, BMAIN_BYTES, threadIdx.x, 128);
    lds_fill_wait();

    int lane  = threadIdx.x & 31;
    int tile  = blockIdx.x * 4 + (threadIdx.x >> 5);
    int b     = tile >> 10;
    int rem   = tile & 1023;
    int h     = rem >> 3;
    int w0    = (rem & 7) << 4;
    int p     = lane & 15;
    int kbase = (lane >> 4) * 8;

    const float* xb   = x + (size_t)b * CHWc;
    const float* offp = off + (size_t)b * 18 * HWc + h * Wc + (w0 + p);

    // preload all 9 taps' offsets so the loads overlap
    float dyk[Kc], dxk[Kc];
#pragma unroll
    for (int k = 0; k < Kc; ++k) {
        dyk[k] = offp[(2 * k) * HWc];
        dxk[k] = offp[(2 * k + 1) * HWc];
    }

    v8f acc[4] = {{}, {}, {}, {}};
    float wq0 = 0, wq1 = 0, wq2 = 0, wq3 = 0;
    int   iq0 = 0, iq1 = 0, iq2 = 0, iq3 = 0;

    for (int t = 0; t < TSTEPS; ++t) {
        if ((t & 1) == 0) {
            int k = t >> 1;
            float py = (float)(h + (k / 3) - 1) + dyk[k];
            float px = (float)(w0 + p + (k % 3) - 1) + dxk[k];
            float y0f = floorf(py), x0f = floorf(px);
            float wy1 = py - y0f,   wx1 = px - x0f;
            float wy0 = 1.0f - wy1, wx0 = 1.0f - wx1;
            int y0 = (int)y0f, x0 = (int)x0f;
            int y1 = y0 + 1,   x1 = x0 + 1;
            bool vy0 = (y0 >= 0) & (y0 < Hc), vy1 = (y1 >= 0) & (y1 < Hc);
            bool vx0 = (x0 >= 0) & (x0 < Wc), vx1 = (x1 >= 0) & (x1 < Wc);
            int y0c = min(max(y0, 0), Hc - 1), y1c = min(max(y1, 0), Hc - 1);
            int x0c = min(max(x0, 0), Wc - 1), x1c = min(max(x1, 0), Wc - 1);
            iq0 = y0c * Wc + x0c;  wq0 = (vy0 & vx0) ? wy0 * wx0 : 0.0f;
            iq1 = y0c * Wc + x1c;  wq1 = (vy0 & vx1) ? wy0 * wx1 : 0.0f;
            iq2 = y1c * Wc + x0c;  wq2 = (vy1 & vx0) ? wy1 * wx0 : 0.0f;
            iq3 = y1c * Wc + x1c;  wq3 = (vy1 & vx1) ? wy1 * wx1 : 0.0f;
        }
        int cbase = (t & 1) * 32;

        V16 ah, al;
#pragma unroll
        for (int j = 0; j < 16; ++j) {
            int Kp = kbase + (j < 8 ? j : j + 8);
            const float* xc = xb + (cbase + Kp) * HWc;
            float v = wq0 * xc[iq0] + wq1 * xc[iq1] + wq2 * xc[iq2] + wq3 * xc[iq3];
            unsigned short hbit = f2bf_rne(v);
            ah.u[j] = hbit;
            al.u[j] = f2bf_rne(v - bf2f(hbit));
        }
#pragma unroll
        for (int nt = 0; nt < 4; ++nt) {
            const v16bf* bph = (const v16bf*)(lds_bh + (((((t * 4 + nt) * 32) + lane) << 4) << 1));
            const v16bf* bpl = (const v16bf*)(lds_bl + (((((t * 4 + nt) * 32) + lane) << 4) << 1));
            v16bf bhv = *bph;
            v16bf blv = *bpl;
            acc[nt] = __builtin_amdgcn_wmma_f32_16x16x32_bf16(false, ah.v, false, bhv, (short)0, acc[nt], false, false);
            acc[nt] = __builtin_amdgcn_wmma_f32_16x16x32_bf16(false, ah.v, false, blv, (short)0, acc[nt], false, false);
            acc[nt] = __builtin_amdgcn_wmma_f32_16x16x32_bf16(false, al.v, false, bhv, (short)0, acc[nt], false, false);
        }
    }

    int n  = lane & 15;
    int mh = (lane >> 4) * 8;
#pragma unroll
    for (int nt = 0; nt < 4; ++nt) {
        int ch = nt * 16 + n;
        float s = sc[ch];
        float o = bi[ch];
        float* op = out + ((size_t)(b * Cc + ch)) * HWc + h * Wc + w0;
#pragma unroll
        for (int r = 0; r < 8; ++r) {
            float val = acc[nt][r] * s + o;
            op[mh + r] = fmaxf(val, 0.0f);
        }
    }
}

// ---------------------------------------------------------------------------
extern "C" void kernel_launch(void* const* d_in, const int* in_sizes, int n_in,
                              void* d_out, int out_size, void* d_ws, size_t ws_size,
                              hipStream_t stream) {
    const float* x      = (const float*)d_in[0];
    const float* w_off1 = (const float*)d_in[1];
    const float* b_off1 = (const float*)d_in[2];
    const float* w1     = (const float*)d_in[3];
    const float* g1     = (const float*)d_in[4];
    const float* be1    = (const float*)d_in[5];
    const float* m1     = (const float*)d_in[6];
    const float* v1     = (const float*)d_in[7];
    const float* w_off2 = (const float*)d_in[8];
    const float* b_off2 = (const float*)d_in[9];
    const float* w2     = (const float*)d_in[10];
    const float* g2     = (const float*)d_in[11];
    const float* be2    = (const float*)d_in[12];
    const float* m2     = (const float*)d_in[13];
    const float* v2     = (const float*)d_in[14];

    char* ws = (char*)d_ws;
    // scratch layout (all 256B aligned)
    float*          off1 = (float*)(ws + 0);                       // 4,718,592 B
    float*          off2 = (float*)(ws + 4718592);                 // 4,718,592 B
    float*          hbuf = (float*)(ws + 9437184);                 // 16,777,216 B
    unsigned short* B1h  = (unsigned short*)(ws + 26214400);       // 73,728 B each
    unsigned short* B1l  = (unsigned short*)(ws + 26288128);
    unsigned short* B2h  = (unsigned short*)(ws + 26361856);
    unsigned short* B2l  = (unsigned short*)(ws + 26435584);
    unsigned short* Bo1h = (unsigned short*)(ws + 26509312);       // 36,864 B each
    unsigned short* Bo1l = (unsigned short*)(ws + 26546176);
    unsigned short* Bo2h = (unsigned short*)(ws + 26583040);
    unsigned short* Bo2l = (unsigned short*)(ws + 26619904);
    float*          sc1  = (float*)(ws + 26656768);
    float*          bi1  = (float*)(ws + 26657024);
    float*          sc2  = (float*)(ws + 26657280);
    float*          bi2  = (float*)(ws + 26657536);

    // weight / BN prep
    pack_w_main<<<144, 256, 0, stream>>>(w1, B1h, B1l);
    pack_w_main<<<144, 256, 0, stream>>>(w2, B2h, B2l);
    pack_w_off <<< 72, 256, 0, stream>>>(w_off1, Bo1h, Bo1l);
    pack_w_off <<< 72, 256, 0, stream>>>(w_off2, Bo2h, Bo2l);
    prep_bn    <<<  1,  64, 0, stream>>>(g1, be1, m1, v1, sc1, bi1);
    prep_bn    <<<  1,  64, 0, stream>>>(g2, be2, m2, v2, sc2, bi2);

    // layer 1
    offconv_kernel<<<1024, 128, 2 * BOFF_BYTES,  stream>>>(x, Bo1h, Bo1l, b_off1, off1);
    dcn_kernel    <<<1024, 128, 2 * BMAIN_BYTES, stream>>>(x, off1, B1h, B1l, sc1, bi1, hbuf);
    // layer 2
    offconv_kernel<<<1024, 128, 2 * BOFF_BYTES,  stream>>>(hbuf, Bo2h, Bo2l, b_off2, off2);
    dcn_kernel    <<<1024, 128, 2 * BMAIN_BYTES, stream>>>(hbuf, off2, B2h, B2l, sc2, bi2, (float*)d_out);
}
